// TransformerEncoder_76845554860243
// MI455X (gfx1250) — matrix operand order repaired and checked by reading, not compiled, not run on hardware
//
#include <hip/hip_runtime.h>
#include <hip/hip_bf16.h>
#include <math.h>

// Problem constants (match reference)
#define BQ   2
#define SQ   2048
#define DQ   512
#define HQ   8
#define LQ   6
#define FQ   2048
#define HDQ  64
#define MAXQ 5000
#define OFFQ (MAXQ - 1)
#define TABQ (2 * MAXQ - 1)

typedef __attribute__((ext_vector_type(16))) __bf16 v16bf;
typedef __attribute__((ext_vector_type(8)))  __bf16 v8bf;
typedef __attribute__((ext_vector_type(8)))  float  v8f;
typedef __attribute__((ext_vector_type(4)))  unsigned int u32x4;
typedef __attribute__((ext_vector_type(4)))  int i32x4;
typedef __attribute__((ext_vector_type(8)))  int i32x8;

static __device__ __forceinline__ v16bf cat16(v8bf lo, v8bf hi) {
  v16bf r;
#pragma unroll
  for (int i = 0; i < 8; ++i) { r[i] = lo[i]; r[i + 8] = hi[i]; }
  return r;
}

// ---------------------------------------------------------------------------
// TDM: async DMA of one 2D bf16 tile (rows x 64B) global -> LDS, with native
// LDS padding of 4 dwords after every 16 dwords (matches LDP=40 bf16 stride).
// D# layout per CDNA5 ISA ch.8 (group0: count/lds_addr/global_addr/type=2;
// group1: data_size=4B units, pad_enable, dims/strides).
// ---------------------------------------------------------------------------
static __device__ __forceinline__ void tdm_load_tile(unsigned lds_addr,
                                                     const __bf16* gptr,
                                                     unsigned strideU,  // row stride, 4B units
                                                     unsigned rows) {
  const unsigned long long ga = (unsigned long long)(uintptr_t)gptr;
  u32x4 g0;
  g0[0] = 1u;                                   // count=1 (valid user descriptor)
  g0[1] = lds_addr;                             // LDS byte address
  g0[2] = (unsigned)ga;                         // global_addr[31:0]
  g0[3] = ((unsigned)(ga >> 32) & 0x1FFFFFFu) | 0x80000000u;  // addr[56:32] | type=2
  i32x8 g1;
  // data_size=2 (4B) | pad_enable | pad_interval=3 (16 dw) | pad_amount=3 (4 dw)
  g1[0] = (int)((2u << 16) | (1u << 20) | (3u << 22) | (3u << 25));
  g1[1] = (int)((strideU & 0xFFFFu) << 16);     // tensor_dim0[15:0]
  g1[2] = (int)((strideU >> 16) & 0xFFFFu);     // tensor_dim0[31:16] | tensor_dim1 lo16=0
  g1[3] = (int)(0x10u | (16u << 16));           // tensor_dim1=1<<20 | tile_dim0=16 units
  g1[4] = (int)(rows & 0xFFFFu);                // tile_dim1=rows, tile_dim2=0
  g1[5] = (int)strideU;                         // tensor_dim0_stride[31:0]
  g1[6] = 0;
  g1[7] = 0;
  const i32x4 z4 = {0, 0, 0, 0};
#if __clang_major__ >= 23
  const i32x8 z8 = {0, 0, 0, 0, 0, 0, 0, 0};
  __builtin_amdgcn_tensor_load_to_lds(g0, g1, z4, z4, z8, 0);
#else
  __builtin_amdgcn_tensor_load_to_lds(g0, g1, z4, z4, 0);
#endif
}

// ---------------------------------------------------------------------------
// Weight convert + transpose: out[N][K] bf16 = in[K][N] fp32 (once per call)
// ---------------------------------------------------------------------------
__global__ void convw_k(const float* __restrict__ in, __bf16* __restrict__ out,
                        int K, int N) {
  const size_t idx = (size_t)blockIdx.x * 256 + threadIdx.x;
  if (idx >= (size_t)K * N) return;
  const int n = (int)(idx / K), k = (int)(idx % K);
  out[idx] = (__bf16)in[(size_t)k * N + n];
}

// ---------------------------------------------------------------------------
// Speaker embedding projection: spkp[b,d] = spk[b,:] @ Wspk[:,d] + bspk[d]
// ---------------------------------------------------------------------------
__global__ void spkproj_k(const float* __restrict__ spk,
                          const float* __restrict__ Wspk,
                          const float* __restrict__ bspk,
                          float* __restrict__ out) {
  const int b = blockIdx.x, d = threadIdx.x;
  float acc = bspk[d];
  for (int k = 0; k < DQ; ++k) acc += spk[b * DQ + k] * Wspk[(size_t)k * DQ + d];
  out[b * DQ + d] = acc;
}

// ---------------------------------------------------------------------------
// Conditional positional encoding (all outer-product adds fused, eval mode)
// ---------------------------------------------------------------------------
__global__ void prep_k(const float* __restrict__ x,  const float* __restrict__ f0,
                       const float* __restrict__ sp, const float* __restrict__ ap,
                       const float* __restrict__ Wf0, const float* __restrict__ bf0,
                       const float* __restrict__ Wsp, const float* __restrict__ bsp,
                       const float* __restrict__ Wap, const float* __restrict__ bap,
                       const float* __restrict__ spkp, float* __restrict__ out) {
  const size_t idx = (size_t)blockIdx.x * blockDim.x + threadIdx.x;
  if (idx >= (size_t)BQ * SQ * DQ) return;
  const int d = (int)(idx % DQ);
  const size_t bs = idx / DQ;
  const int s = (int)(bs % SQ);
  const int b = (int)(bs / SQ);
  const float div = __expf((float)(d & ~1) * (-9.210340371976184f / (float)DQ));
  const float pe = (d & 1) ? __cosf((float)s * div) : __sinf((float)s * div);
  out[idx] = x[idx] + pe
           + f0[bs] * Wf0[d] + bf0[d]
           + sp[bs] * Wsp[d] + bsp[d]
           + ap[bs] * Wap[d] + bap[d]
           + spkp[b * DQ + d];
}

// ---------------------------------------------------------------------------
// LayerNorm over D=512; writes fp32 (outF) or bf16 (outB) rows
// ---------------------------------------------------------------------------
__global__ __launch_bounds__(128) void layernorm_k(const float* __restrict__ in,
                                                   const float* __restrict__ g,
                                                   const float* __restrict__ b,
                                                   float* __restrict__ outF,
                                                   __bf16* __restrict__ outB) {
  const int row = blockIdx.x, t = threadIdx.x;
  const float* x = in + (size_t)row * DQ;
  float v[4], s = 0.f, ss = 0.f;
#pragma unroll
  for (int j = 0; j < 4; ++j) { v[j] = x[t + 128 * j]; s += v[j]; ss += v[j] * v[j]; }
#pragma unroll
  for (int m = 16; m >= 1; m >>= 1) { s += __shfl_xor(s, m, 32); ss += __shfl_xor(ss, m, 32); }
  __shared__ float sh[8];
  const int w = t >> 5, lane = t & 31;
  if (lane == 0) { sh[w] = s; sh[4 + w] = ss; }
  __syncthreads();
  s  = sh[0] + sh[1] + sh[2] + sh[3];
  ss = sh[4] + sh[5] + sh[6] + sh[7];
  const float mean = s * (1.f / DQ);
  const float inv  = rsqrtf(ss * (1.f / DQ) - mean * mean + 1e-5f);
#pragma unroll
  for (int j = 0; j < 4; ++j) {
    const int d = t + 128 * j;
    const float r = (v[j] - mean) * inv * g[d] + b[d];
    if (outF) outF[(size_t)row * DQ + d] = r;
    else      outB[(size_t)row * DQ + d] = (__bf16)r;
  }
}

// ---------------------------------------------------------------------------
// GEMM: out[M,N] = act(A[M,K] @ Wt[N,K]^T + bias) (+ residual)
// A bf16 row-major, Wt bf16 pre-transposed [N][K]. Block tile 128x64, BK=32,
// 8 waves (4M x 2N), 32x32 per wave: 2 A-frags x 2 B-frags -> 4 WMMAs/K-step
// (2 ds_load_b128 per WMMA, register-level fragment reuse). Double-buffered
// LDS tiles filled by the Tensor Data Mover; s_wait_tensorcnt overlaps the
// in-flight DMA with the current tile's WMMAs.
// ---------------------------------------------------------------------------
#define BM  128
#define BN  64
#define BK  32
#define LDP 40  // padded LDS row stride (bf16); TDM pad 16+4 dwords reproduces it

__global__ __launch_bounds__(256) void gemm_bf16_wmma(
    const __bf16* __restrict__ A, const __bf16* __restrict__ Wt,
    const float* __restrict__ bias, const float* __restrict__ resid,
    float* __restrict__ outF, __bf16* __restrict__ outB,
    int M, int N, int K, int doGelu) {
  __shared__ __attribute__((aligned(16))) __bf16 tiles[2][(BM + BN) * LDP];
  const int t = threadIdx.x;
  const int lane = t & 31, half = lane >> 4, l15 = lane & 15;
  const int wave = t >> 5, wm = wave & 3, wn = wave >> 2;
  const int m0 = blockIdx.y * BM, n0 = blockIdx.x * BN;
  const unsigned strideU = (unsigned)(K >> 1);  // row stride in 4B units

  const v8f zero = {0, 0, 0, 0, 0, 0, 0, 0};
  v8f acc[2][2] = {{zero, zero}, {zero, zero}};  // [rowGroup][colGroup]

  // Preload tile 0 via TDM
  if (wave == 0)
    tdm_load_tile((unsigned)(uintptr_t)&tiles[0][0], A + (size_t)m0 * K, strideU, BM);
  else if (wave == 1)
    tdm_load_tile((unsigned)(uintptr_t)&tiles[0][BM * LDP], Wt + (size_t)n0 * K, strideU, BN);

  const int nsteps = K / BK;
  for (int s = 0; s < nsteps; ++s) {
    const int cur = s & 1;
    if (s + 1 < nsteps) {  // issue next tile into the other buffer, then wait for current
      const int k1 = (s + 1) * BK;
      if (wave == 0)
        tdm_load_tile((unsigned)(uintptr_t)&tiles[cur ^ 1][0],
                      A + (size_t)m0 * K + k1, strideU, BM);
      else if (wave == 1)
        tdm_load_tile((unsigned)(uintptr_t)&tiles[cur ^ 1][BM * LDP],
                      Wt + (size_t)n0 * K + k1, strideU, BN);
      __builtin_amdgcn_s_wait_tensorcnt(1);
    } else {
      __builtin_amdgcn_s_wait_tensorcnt(0);
    }
    __syncthreads();

    const __bf16* As = &tiles[cur][0];
    const __bf16* Bs = &tiles[cur][BM * LDP];
    // A fragments (ISA 16-bit A layout): row m, chunks at half*8 and 16+half*8
    v16bf af[2];
#pragma unroll
    for (int r = 0; r < 2; ++r) {
      const int a0 = (wm * 32 + r * 16 + l15) * LDP + half * 8;
      af[r] = cat16(*(const v8bf*)&As[a0], *(const v8bf*)&As[a0 + 16]);
    }
    // B fragments: lane holds col n, chunks at half*8 and 16+half*8 (same K order as A)
    v16bf bf[2];
#pragma unroll
    for (int cgl = 0; cgl < 2; ++cgl) {
      const int b0 = (wn * 32 + cgl * 16 + l15) * LDP + half * 8;
      bf[cgl] = cat16(*(const v8bf*)&Bs[b0], *(const v8bf*)&Bs[b0 + 16]);
    }
#pragma unroll
    for (int r = 0; r < 2; ++r)
#pragma unroll
      for (int cgl = 0; cgl < 2; ++cgl)
        acc[r][cgl] = __builtin_amdgcn_wmma_f32_16x16x32_bf16(
            false, af[r], false, bf[cgl], (short)0, acc[r][cgl], false, false);
    __syncthreads();
  }

  // Epilogue: bias (+gelu) (+residual). C layout: vgpr i -> row i (+8 for half 1), col = l15.
#pragma unroll
  for (int r = 0; r < 2; ++r) {
#pragma unroll
    for (int cgl = 0; cgl < 2; ++cgl) {
      const v8f c = acc[r][cgl];
      const int col = n0 + wn * 32 + cgl * 16 + l15;
#pragma unroll
      for (int i = 0; i < 8; ++i) {
        const int row = m0 + wm * 32 + r * 16 + half * 8 + i;
        float v = c[i] + bias[col];
        if (doGelu) v = 0.5f * v * (1.0f + erff(v * 0.70710678118654752f));
        if (resid) v += resid[(size_t)row * N + col];
        if (outF) outF[(size_t)row * N + col] = v;
        else      outB[(size_t)row * N + col] = (__bf16)v;
      }
    }
  }
}

// ---------------------------------------------------------------------------
// Flash-style attention with relative-position bias (bf16 in, bf16 out).
// Block = 128 threads (4 waves); each wave owns 16 query rows; block covers a
// 64-row query tile of one (batch, head). KV consumed in chunks of 32 keys.
// ---------------------------------------------------------------------------
#define VP 40
__global__ __launch_bounds__(128) void attn_wmma(
    const __bf16* __restrict__ qb, const __bf16* __restrict__ kb,
    const __bf16* __restrict__ vb, const float* __restrict__ tab,
    __bf16* __restrict__ ao) {
  __shared__ __attribute__((aligned(16))) __bf16 Vt[HDQ * VP];     // [hd][key] transposed V chunk
  __shared__ __attribute__((aligned(16))) __bf16 Pw[4 * 16 * VP];  // per-wave 16x32 prob tile
  const int t = threadIdx.x;
  const int lane = t & 31, half = lane >> 4, l15 = lane & 15;
  const int wave = t >> 5;
  const int q0w = blockIdx.x * 64 + wave * 16;
  const int hh = blockIdx.y, bb = blockIdx.z;
  const float scale = 0.125f;  // 1/sqrt(64)

  // Q fragments (persist whole kernel): Q rows are already in A-fragment layout
  const __bf16* qrow = qb + ((size_t)(bb * SQ + q0w + l15)) * DQ + hh * HDQ;
  v16bf qf[2];
#pragma unroll
  for (int kt = 0; kt < 2; ++kt) {
    const int kk = kt * 32 + half * 8;
    qf[kt] = cat16(*(const v8bf*)(qrow + kk), *(const v8bf*)(qrow + kk + 16));
  }

  const v8f zero = {0, 0, 0, 0, 0, 0, 0, 0};
  float m_run[8], l_run[8];
  v8f o[4];
#pragma unroll
  for (int i = 0; i < 8; ++i) { m_run[i] = -3.0e38f; l_run[i] = 0.f; }
#pragma unroll
  for (int nt = 0; nt < 4; ++nt) o[nt] = zero;

  __bf16* pw = &Pw[wave * 16 * VP];

  for (int k0c = 0; k0c < SQ; k0c += 32) {
    // Stage V chunk transposed: vectorized 16B reads, scatter to Vt[hd][key]
    for (int c = t; c < 256; c += 128) {
      const int key = c >> 3, hd0 = (c & 7) * 8;
      const v8bf vv = *(const v8bf*)(vb + ((size_t)(bb * SQ + k0c + key)) * DQ + hh * HDQ + hd0);
#pragma unroll
      for (int j = 0; j < 8; ++j) Vt[(hd0 + j) * VP + key] = vv[j];
    }
    __syncthreads();

    // Scores: S[q, key] = Q @ K^T  (K rows in global are exactly the B-fragment slices)
    v8f s0 = zero, s1 = zero;
#pragma unroll
    for (int nt = 0; nt < 2; ++nt) {
      const int key = k0c + nt * 16 + l15;
      const __bf16* krow = kb + ((size_t)(bb * SQ + key)) * DQ + hh * HDQ;
#pragma unroll
      for (int kt = 0; kt < 2; ++kt) {
        const int kk = kt * 32 + half * 8;
        const v16bf kf = cat16(*(const v8bf*)(krow + kk), *(const v8bf*)(krow + kk + 16));
        if (nt == 0)
          s0 = __builtin_amdgcn_wmma_f32_16x16x32_bf16(false, qf[kt], false, kf, (short)0, s0, false, false);
        else
          s1 = __builtin_amdgcn_wmma_f32_16x16x32_bf16(false, qf[kt], false, kf, (short)0, s1, false, false);
      }
    }

    // Relative bias + online softmax (row = vgpr i + half*8, col = l15 in C layout)
    float alpha[8];
#pragma unroll
    for (int i = 0; i < 8; ++i) {
      const int qr = q0w + half * 8 + i;
      int i0 = k0c + l15 - qr + OFFQ;       i0 = i0 < 0 ? 0 : (i0 > TABQ - 1 ? TABQ - 1 : i0);
      int i1 = k0c + 16 + l15 - qr + OFFQ;  i1 = i1 < 0 ? 0 : (i1 > TABQ - 1 ? TABQ - 1 : i1);
      const float v0 = s0[i] * scale + tab[(size_t)i0 * HQ + hh];
      const float v1 = s1[i] * scale + tab[(size_t)i1 * HQ + hh];
      float mx = fmaxf(v0, v1);
#pragma unroll
      for (int m = 8; m >= 1; m >>= 1) mx = fmaxf(mx, __shfl_xor(mx, m, 16));
      const float mnew = fmaxf(m_run[i], mx);
      alpha[i] = __expf(m_run[i] - mnew);
      const float p0 = __expf(v0 - mnew), p1 = __expf(v1 - mnew);
      float rs = p0 + p1;
#pragma unroll
      for (int m = 8; m >= 1; m >>= 1) rs += __shfl_xor(rs, m, 16);
      l_run[i] = l_run[i] * alpha[i] + rs;
      m_run[i] = mnew;
      // C layout -> row-major LDS (reformat for A-fragment reload)
      pw[(half * 8 + i) * VP + l15]      = (__bf16)p0;
      pw[(half * 8 + i) * VP + 16 + l15] = (__bf16)p1;
    }
    __syncthreads();

    // P fragment (A layout) and PV accumulation against transposed V
    const int pr = l15 * VP + half * 8;
    const v16bf pf = cat16(*(const v8bf*)&pw[pr], *(const v8bf*)&pw[pr + 16]);
#pragma unroll
    for (int nt = 0; nt < 4; ++nt) {
#pragma unroll
      for (int i = 0; i < 8; ++i) o[nt][i] *= alpha[i];
      const int vf0 = (nt * 16 + l15) * VP + half * 8;
      const v16bf vf = cat16(*(const v8bf*)&Vt[vf0], *(const v8bf*)&Vt[vf0 + 16]);
      o[nt] = __builtin_amdgcn_wmma_f32_16x16x32_bf16(false, pf, false, vf, (short)0, o[nt], false, false);
    }
    __syncthreads();
  }

  // Normalize and write (B,S,H*HD) bf16 (consumed by O-projection GEMM)
#pragma unroll
  for (int nt = 0; nt < 4; ++nt) {
#pragma unroll
    for (int i = 0; i < 8; ++i) {
      const int row = q0w + half * 8 + i;
      const int hd = nt * 16 + l15;
      ao[((size_t)(bb * SQ + row)) * DQ + hh * HDQ + hd] = (__bf16)(o[nt][i] / l_run[i]);
    }
  }
}

// ---------------------------------------------------------------------------
// Host-side orchestration
// ---------------------------------------------------------------------------
extern "C" void kernel_launch(void* const* d_in, const int* in_sizes, int n_in,
                              void* d_out, int out_size, void* d_ws, size_t ws_size,
                              hipStream_t stream) {
  (void)in_sizes; (void)n_in; (void)out_size; (void)ws_size;
  const float* x    = (const float*)d_in[0];
  const float* f0   = (const float*)d_in[1];
  const float* sp   = (const float*)d_in[2];
  const float* ap   = (const float*)d_in[3];
  const float* spk  = (const float*)d_in[4];
  const float* Wf0  = (const float*)d_in[5];
  const float* bf0  = (const float*)d_in[6];
  const float* Wsp  = (const float*)d_in[7];
  const float* bsp  = (const float*)d_in[8];
  const float* Wap  = (const float*)d_in[9];
  const float* bap  = (const float*)d_in[10];
  const float* Wspk = (const float*)d_in[11];
  const float* bspk = (const float*)d_in[12];
  const float* Wq   = (const float*)d_in[13];
  const float* bq   = (const float*)d_in[14];
  const float* Wk   = (const float*)d_in[15];
  const float* bk   = (const float*)d_in[16];
  const float* Wv   = (const float*)d_in[17];
  const float* bv   = (const float*)d_in[18];
  const float* Wo   = (const float*)d_in[19];
  const float* bo   = (const float*)d_in[20];
  const float* g1   = (const float*)d_in[21];
  const float* be1  = (const float*)d_in[22];
  const float* g2   = (const float*)d_in[23];
  const float* be2  = (const float*)d_in[24];
  const float* W1   = (const float*)d_in[25];
  const float* b1   = (const float*)d_in[26];
  const float* W2   = (const float*)d_in[27];
  const float* b2   = (const float*)d_in[28];
  const float* tab  = (const float*)d_in[29];
  const float* gf   = (const float*)d_in[30];
  const float* bfin = (const float*)d_in[31];

  const int M = BQ * SQ;                       // 4096 rows
  const size_t S0 = (size_t)M * DQ * 4;        // 8 MB activation tensor
  char* ws = (char*)d_ws;
  float*  x_ws = (float*)(ws);                               // residual stream (fp32)
  __bf16* h_ws = (__bf16*)(ws + S0);                         // LN output (bf16)
  __bf16* ao_ws = (__bf16*)(ws + S0 + S0 / 2);               // attention out (bf16)
  // Region R (2*S0 bytes): FFN intermediate bf16 [M,F]; QKV bf16 alias it
  __bf16* t_ws = (__bf16*)(ws + 2 * S0);
  __bf16* qbuf = (__bf16*)(ws + 2 * S0);
  __bf16* kbuf = (__bf16*)(ws + 2 * S0 + S0 / 2);
  __bf16* vbuf = (__bf16*)(ws + 2 * S0 + S0);
  float*  spkp = (float*)(ws + 4 * S0);                      // [B, D]
  __bf16* wbf  = (__bf16*)(ws + 4 * S0 + 4096);              // transposed bf16 weights

  const size_t WDD = (size_t)DQ * DQ;     // 262144
  const size_t WDF = (size_t)DQ * FQ;     // 1048576
  const size_t WLAYER = 4 * WDD + 2 * WDF;

  // One-time (per call) weight convert + transpose to [N][K] bf16
  for (int l = 0; l < LQ; ++l) {
    __bf16* wl = wbf + (size_t)l * WLAYER;
    convw_k<<<(int)(WDD / 256), 256, 0, stream>>>(Wq + l * WDD, wl + 0 * WDD, DQ, DQ);
    convw_k<<<(int)(WDD / 256), 256, 0, stream>>>(Wk + l * WDD, wl + 1 * WDD, DQ, DQ);
    convw_k<<<(int)(WDD / 256), 256, 0, stream>>>(Wv + l * WDD, wl + 2 * WDD, DQ, DQ);
    convw_k<<<(int)(WDD / 256), 256, 0, stream>>>(Wo + l * WDD, wl + 3 * WDD, DQ, DQ);
    convw_k<<<(int)(WDF / 256), 256, 0, stream>>>(W1 + l * WDF, wl + 4 * WDD, DQ, FQ);
    convw_k<<<(int)(WDF / 256), 256, 0, stream>>>(W2 + l * WDF, wl + 4 * WDD + WDF, FQ, DQ);
  }

  spkproj_k<<<BQ, DQ, 0, stream>>>(spk, Wspk, bspk, spkp);
  prep_k<<<(M * DQ) / 256, 256, 0, stream>>>(x, f0, sp, ap, Wf0, bf0, Wsp, bsp,
                                             Wap, bap, spkp, x_ws);

  const dim3 gD(DQ / BN, M / BM);   // N=512 GEMMs  (8 x 32 blocks)
  const dim3 gF(FQ / BN, M / BM);   // N=2048 GEMM  (32 x 32 blocks)
  const dim3 gA(SQ / 64, HQ, BQ);   // attention

  for (int l = 0; l < LQ; ++l) {
    const __bf16* wl = wbf + (size_t)l * WLAYER;
    layernorm_k<<<M, 128, 0, stream>>>(x_ws, g1 + l * DQ, be1 + l * DQ, nullptr, h_ws);
    gemm_bf16_wmma<<<gD, 256, 0, stream>>>(h_ws, wl + 0 * WDD, bq + l * DQ, nullptr,
                                           nullptr, qbuf, M, DQ, DQ, 0);
    gemm_bf16_wmma<<<gD, 256, 0, stream>>>(h_ws, wl + 1 * WDD, bk + l * DQ, nullptr,
                                           nullptr, kbuf, M, DQ, DQ, 0);
    gemm_bf16_wmma<<<gD, 256, 0, stream>>>(h_ws, wl + 2 * WDD, bv + l * DQ, nullptr,
                                           nullptr, vbuf, M, DQ, DQ, 0);
    attn_wmma<<<gA, 128, 0, stream>>>(qbuf, kbuf, vbuf,
                                      tab + (size_t)l * TABQ * HQ, ao_ws);
    gemm_bf16_wmma<<<gD, 256, 0, stream>>>(ao_ws, wl + 3 * WDD, bo + l * DQ, x_ws,
                                           x_ws, nullptr, M, DQ, DQ, 0);
    layernorm_k<<<M, 128, 0, stream>>>(x_ws, g2 + l * DQ, be2 + l * DQ, nullptr, h_ws);
    gemm_bf16_wmma<<<gF, 256, 0, stream>>>(h_ws, wl + 4 * WDD, b1 + l * FQ, nullptr,
                                           nullptr, t_ws, M, FQ, DQ, 1);
    gemm_bf16_wmma<<<gD, 256, 0, stream>>>(t_ws, wl + 4 * WDD + WDF, b2 + l * DQ, x_ws,
                                           x_ws, nullptr, M, DQ, FQ, 0);
  }
  layernorm_k<<<M, 128, 0, stream>>>(x_ws, gf, bfin, (float*)d_out, nullptr);
}
